// DeltaNet_22488448762155
// MI455X (gfx1250) — matrix operand order repaired and checked by reading, not compiled, not run on hardware
//
#include <hip/hip_runtime.h>
#include <hip/hip_bf16.h>
#include <math.h>

// ---------------- problem constants ----------------
#define BB   4
#define SS   2048
#define HH   1024
#define NHH  16
#define DD   64
#define MTOK (BB * SS)          // 8192 token rows
#define EPSV     1e-6f
#define V_CLIPV  2.0f
#define LN_EPSV  1e-5f

// ---------------- WMMA types ----------------
typedef __attribute__((ext_vector_type(16))) __bf16 v16bf;
typedef __attribute__((ext_vector_type(8)))  float  v8f;

struct __attribute__((aligned(16))) U4 { unsigned int x, y, z, w; };
union Frag { U4 q[2]; v16bf v; };                  // two 16B loads -> operand

__device__ __forceinline__ unsigned short f2bf_bits(float f) {
  unsigned u = __builtin_bit_cast(unsigned, f);
  u += 0x7FFFu + ((u >> 16) & 1u);                 // round-to-nearest-even
  return (unsigned short)(u >> 16);
}
__device__ __forceinline__ float bf2f(unsigned short s) {
  return __builtin_bit_cast(float, (unsigned)s << 16);
}

// ---- CDNA5 async global->LDS copy (ASYNCcnt path, 16B per lane) ----
__device__ __forceinline__ void async_b128(const void* gsrc, const void* lds_dst) {
  unsigned loff = (unsigned)(size_t)lds_dst;       // low 32 bits = LDS byte address
  asm volatile("global_load_async_to_lds_b128 %0, %1, off"
               :: "v"(loff), "v"(gsrc) : "memory");
}

// =====================================================================
// Kernel 0: one-shot f32 -> bf16 conversion (x and the four weights).
// Bandwidth-bound pre-pass so NO conversion VALU pollutes the GEMM loops.
// =====================================================================
__global__ __launch_bounds__(256) void cvt_bf16_kernel(
    const float* __restrict__ src, unsigned short* __restrict__ dst, int npairs)
{
  int i = blockIdx.x * blockDim.x + threadIdx.x;
  const int stride = gridDim.x * blockDim.x;
  for (; i < npairs; i += stride) {
    const float f0 = src[2 * i], f1 = src[2 * i + 1];
    ((unsigned*)dst)[i] = (unsigned)f2bf_bits(f0) | ((unsigned)f2bf_bits(f1) << 16);
  }
}

// =====================================================================
// Kernel 1: fused QKV projection on bf16 data.
// Block tile 128Mx64N, 8 waves (4Mx2N), wave 32x32 = 2x2 WMMA tiles.
// Tiles DMA'd into LDS with global_load_async_to_lds_b128 (double
// buffered, s_wait_asynccnt); fragments are raw ds_load_b128.
// =====================================================================
#define TSTRIDE 40   // ushorts; 80B rows (16B aligned), gcd(20,64)=4 -> conflict-free

__global__ __launch_bounds__(256) void qkv_kernel(
    const unsigned short* __restrict__ xb,
    const unsigned short* __restrict__ Wqb, const unsigned short* __restrict__ Wkb,
    const unsigned short* __restrict__ Wvb,
    unsigned short* __restrict__ qo, unsigned short* __restrict__ ko,
    unsigned short* __restrict__ vo)
{
  const int m0 = blockIdx.x * 128;
  const int n0 = blockIdx.y * 64;
  const int which = blockIdx.z;                       // 0=q 1=k 2=v
  const unsigned short* __restrict__ W = (which == 0) ? Wqb : (which == 1) ? Wkb : Wvb;
  unsigned short* __restrict__ out     = (which == 0) ? qo  : (which == 1) ? ko  : vo;

  __shared__ unsigned short Xs[2][128 * TSTRIDE];     // bf16 tiles, double buffered
  __shared__ unsigned short Ws[2][64 * TSTRIDE];

  const int tid  = threadIdx.x;
  const int wave = tid >> 5;
  const int lane = tid & 31;
  const int wm   = wave & 3;                          // M quadrant (32 rows)
  const int wn   = wave >> 2;                         // N half (32 cols)
  const int l16  = lane & 15;
  const int hi   = lane >> 4;                         // lane half

  v8f acc[2][2] = {};

  // X tile: 512 x 16B chunks (2/thr) ; W tile: 256 chunks (1/thr) = 3 async/thr
  auto stage = [&](int kk, int buf) {
    #pragma unroll
    for (int j = 0; j < 2; ++j) {
      int c = tid + 256 * j;
      int r = c >> 2, f = (c & 3) << 3;               // 4 chunks of 8 bf16 per row
      async_b128(xb + (size_t)(m0 + r) * HH + kk + f, &Xs[buf][r * TSTRIDE + f]);
    }
    {
      int r = tid >> 2, f = (tid & 3) << 3;
      async_b128(W + (size_t)(n0 + r) * HH + kk + f, &Ws[buf][r * TSTRIDE + f]);
    }
  };

  stage(0, 0);
  for (int step = 0; step < HH / 32; ++step) {
    const int buf = step & 1;
    if (step + 1 < HH / 32) {
      stage((step + 1) * 32, buf ^ 1);                        // prefetch next tile
      asm volatile("s_wait_asynccnt 0x3" ::: "memory");       // current tile landed
    } else {
      asm volatile("s_wait_asynccnt 0x0" ::: "memory");
    }
    __syncthreads();

    // A (16x32 MxK): lane half -> K 0-15 / 16-31 contiguous
    Frag a0, a1, b0, b1;
    {
      const unsigned short* p = &Xs[buf][(wm * 32 +      l16) * TSTRIDE + hi * 16];
      a0.q[0] = *(const U4*)(p);  a0.q[1] = *(const U4*)(p + 8);
    }
    {
      const unsigned short* p = &Xs[buf][(wm * 32 + 16 + l16) * TSTRIDE + hi * 16];
      a1.q[0] = *(const U4*)(p);  a1.q[1] = *(const U4*)(p + 8);
    }
    // B (32x16 KxN): lane = N col, lane half -> K 0-15 / 16-31 contiguous
    {
      const unsigned short* p = &Ws[buf][(wn * 32 +      l16) * TSTRIDE + hi * 16];
      b0.q[0] = *(const U4*)(p);  b0.q[1] = *(const U4*)(p + 8);
    }
    {
      const unsigned short* p = &Ws[buf][(wn * 32 + 16 + l16) * TSTRIDE + hi * 16];
      b1.q[0] = *(const U4*)(p);  b1.q[1] = *(const U4*)(p + 8);
    }

    acc[0][0] = __builtin_amdgcn_wmma_f32_16x16x32_bf16(false, a0.v, false, b0.v, (short)0, acc[0][0], false, false);
    acc[0][1] = __builtin_amdgcn_wmma_f32_16x16x32_bf16(false, a0.v, false, b1.v, (short)0, acc[0][1], false, false);
    acc[1][0] = __builtin_amdgcn_wmma_f32_16x16x32_bf16(false, a1.v, false, b0.v, (short)0, acc[1][0], false, false);
    acc[1][1] = __builtin_amdgcn_wmma_f32_16x16x32_bf16(false, a1.v, false, b1.v, (short)0, acc[1][1], false, false);
    __syncthreads();                                  // reads done before overwrite
  }

  // epilogue: C element i -> m = i + 8*(lane>>4), n = lane&15 ; branchless phi
  #pragma unroll
  for (int mt = 0; mt < 2; ++mt)
    #pragma unroll
    for (int nt = 0; nt < 2; ++nt)
      #pragma unroll
      for (int i = 0; i < 8; ++i) {
        int m = m0 + wm * 32 + mt * 16 + i + 8 * hi;
        int n = n0 + wn * 32 + nt * 16 + l16;
        float z = acc[mt][nt][i];
        if (which < 2) {
          float ez = __expf(z);                       // phi = elu+1, select not branch
          z = (z > 0.f) ? (z + 1.f) : ez;
        }
        out[(size_t)m * HH + n] = f2bf_bits(z);
      }
}

// =====================================================================
// Kernel 2: sequential DeltaNet scan — ONE wave32 per (b,h).
// Lane owns state columns e=lane, e=lane+32 (128 VGPRs) + svec elems.
// All reductions / broadcasts are wave shuffles: zero barriers, zero LDS.
// =====================================================================
__global__ __launch_bounds__(32) void scan_kernel(
    const unsigned short* __restrict__ pq, const unsigned short* __restrict__ pk,
    const unsigned short* __restrict__ v,  const int*  __restrict__ mask,
    const float* __restrict__ theta, const float* __restrict__ gate_w,
    const float* __restrict__ gate_b, unsigned short* __restrict__ y)
{
  const int bh = blockIdx.x;
  const int b  = bh / NHH, h = bh % NHH;
  const int t  = threadIdx.x;                         // lane 0..31

  const float th  = theta[h];
  const float sp  = (th > 20.f) ? th : log1pf(__expf(th));   // softplus
  const float lam = __expf(-sp);
  const float gwq0 = gate_w[h * 2 * DD + t];
  const float gwq1 = gate_w[h * 2 * DD + t + 32];
  const float gwk0 = gate_w[h * 2 * DD + DD + t];
  const float gwk1 = gate_w[h * 2 * DD + DD + t + 32];
  const float gb   = gate_b[h];

  float S0[64], S1[64];
  #pragma unroll
  for (int d = 0; d < 64; ++d) { S0[d] = 0.f; S1[d] = 0.f; }
  float sv0 = 0.f, sv1 = 0.f;

  for (int s = 0; s < SS; ++s) {
    const size_t base = (size_t)(b * SS + s) * HH + h * DD;
    const float pq0 = bf2f(pq[base + t]), pq1 = bf2f(pq[base + t + 32]);
    const float pk0 = bf2f(pk[base + t]), pk1 = bf2f(pk[base + t + 32]);
    const float v0  = bf2f(v[base + t]),  v1  = bf2f(v[base + t + 32]);

    // fused reduction: ||v||^2 and gate logit (5 shuffle levels, no barriers)
    float rv = v0 * v0 + v1 * v1;
    float rg = pq0 * gwq0 + pk0 * gwk0 + pq1 * gwq1 + pk1 * gwk1;
    #pragma unroll
    for (int off = 16; off > 0; off >>= 1) {
      rv += __shfl_xor(rv, off, 32);
      rg += __shfl_xor(rg, off, 32);
    }
    const float mk    = (float)mask[b * SS + s];
    const float scale = fminf(V_CLIPV / (sqrtf(rv) + EPSV), 1.f);
    const float g     = mk / (1.f + __expf(-(rg + gb)));

    sv0 = sv0 * lam + g * pk0;
    sv1 = sv1 * lam + g * pk1;
    const float gv0 = g * (v0 * scale);
    const float gv1 = g * (v1 * scale);

    float n0 = 0.f, n1 = 0.f;
    #pragma unroll
    for (int d = 0; d < 32; ++d) {                    // pk[d], pq[d] via lane bcast
      const float pkd = __shfl(pk0, d, 32);
      const float pqd = __shfl(pq0, d, 32);
      S0[d] = S0[d] * lam + pkd * gv0;  n0 += pqd * S0[d];
      S1[d] = S1[d] * lam + pkd * gv1;  n1 += pqd * S1[d];
    }
    #pragma unroll
    for (int d = 0; d < 32; ++d) {
      const float pkd = __shfl(pk1, d, 32);
      const float pqd = __shfl(pq1, d, 32);
      S0[d + 32] = S0[d + 32] * lam + pkd * gv0;  n0 += pqd * S0[d + 32];
      S1[d + 32] = S1[d + 32] * lam + pkd * gv1;  n1 += pqd * S1[d + 32];
    }

    float rd = pq0 * sv0 + pq1 * sv1;                 // denom (post-update svec)
    #pragma unroll
    for (int off = 16; off > 0; off >>= 1) rd += __shfl_xor(rd, off, 32);
    const float inv = 1.f / (rd + EPSV);
    y[base + t]      = f2bf_bits(n0 * inv);
    y[base + t + 32] = f2bf_bits(n1 * inv);
  }
}

// =====================================================================
// Kernel 3: out = LayerNorm(x + y @ Wo^T + bo).  Block = full 16x1024 row
// slab so LN fuses; 8 waves x (16x128).  y(bf16) A-tile async-DMA'd to
// LDS; Wo(bf16, 2MB L2-resident) B-fragments as raw b128 global loads.
// =====================================================================
__global__ __launch_bounds__(256) void out_ln_kernel(
    const unsigned short* __restrict__ y, const unsigned short* __restrict__ Wo,
    const float* __restrict__ bo, const float* __restrict__ x,
    const float* __restrict__ gamma, const float* __restrict__ beta,
    float* __restrict__ out)
{
  const int m0   = blockIdx.x * 16;
  const int tid  = threadIdx.x;
  const int wave = tid >> 5, lane = tid & 31;
  const int l16  = lane & 15, hi = lane >> 4;
  const int n0w  = wave * 128;

  __shared__ unsigned short Ys[2][16 * TSTRIDE];      // 16x32 bf16, double buffered
  __shared__ float rsum[16], rsumsq[16];

  v8f acc[8] = {};

  auto stageY = [&](int kk, int buf) {                // 64 chunks: tid<64 issue 1 each
    if (tid < 64) {
      int r = tid >> 2, f = (tid & 3) << 3;
      async_b128(y + (size_t)(m0 + r) * HH + kk + f, &Ys[buf][r * TSTRIDE + f]);
    }
  };

  stageY(0, 0);
  for (int step = 0; step < HH / 32; ++step) {
    const int kk = step * 32;
    const int buf = step & 1;
    if (step + 1 < HH / 32) {
      stageY(kk + 32, buf ^ 1);
      asm volatile("s_wait_asynccnt 0x1" ::: "memory");
    } else {
      asm volatile("s_wait_asynccnt 0x0" ::: "memory");
    }
    __syncthreads();

    Frag a;
    const unsigned short* p = &Ys[buf][l16 * TSTRIDE + hi * 16];
    a.q[0] = *(const U4*)(p);
    a.q[1] = *(const U4*)(p + 8);
    #pragma unroll
    for (int nt = 0; nt < 8; ++nt) {
      const int n = n0w + nt * 16 + l16;
      const unsigned short* wp = Wo + (size_t)n * HH + kk + hi * 16;
      Frag bf;
      bf.q[0] = *(const U4*)(wp);                     // 16 contiguous bf16 K-values
      bf.q[1] = *(const U4*)(wp + 8);
      acc[nt] = __builtin_amdgcn_wmma_f32_16x16x32_bf16(
          false, a.v, false, bf.v, (short)0, acc[nt], false, false);
    }
    __syncthreads();
  }

  if (tid < 16) { rsum[tid] = 0.f; rsumsq[tid] = 0.f; }
  __syncthreads();

  #pragma unroll
  for (int nt = 0; nt < 8; ++nt)
    #pragma unroll
    for (int i = 0; i < 8; ++i) {
      const int m = i + 8 * hi;
      const int n = n0w + nt * 16 + l16;
      float zv = acc[nt][i] + bo[n] + x[(size_t)(m0 + m) * HH + n];  // bias + residual
      acc[nt][i] = zv;
      atomicAdd(&rsum[m],   zv);
      atomicAdd(&rsumsq[m], zv * zv);
    }
  __syncthreads();

  #pragma unroll
  for (int nt = 0; nt < 8; ++nt)
    #pragma unroll
    for (int i = 0; i < 8; ++i) {
      const int m = i + 8 * hi;
      const int n = n0w + nt * 16 + l16;
      const float mu  = rsum[m] * (1.f / HH);
      const float var = rsumsq[m] * (1.f / HH) - mu * mu;
      const float zv  = acc[nt][i];
      out[(size_t)(m0 + m) * HH + n] =
          (zv - mu) * rsqrtf(var + LN_EPSV) * gamma[n] + beta[n];
    }
}

// =====================================================================
extern "C" void kernel_launch(void* const* d_in, const int* in_sizes, int n_in,
                              void* d_out, int out_size, void* d_ws, size_t ws_size,
                              hipStream_t stream) {
  const float* x       = (const float*)d_in[0];
  const int*   mask    = (const int*)  d_in[1];
  const float* Wq      = (const float*)d_in[2];
  const float* Wk      = (const float*)d_in[3];
  const float* Wv      = (const float*)d_in[4];
  const float* Wo      = (const float*)d_in[5];
  const float* bo      = (const float*)d_in[6];
  const float* theta   = (const float*)d_in[7];
  const float* gate_w  = (const float*)d_in[8];
  const float* gate_b  = (const float*)d_in[9];
  const float* gamma   = (const float*)d_in[10];
  const float* beta    = (const float*)d_in[11];

  // workspace: all-bf16 intermediates (q,k,v,y planes + x + 4 weights) = 92MB
  unsigned short* u = (unsigned short*)d_ws;
  const size_t plane = (size_t)MTOK * HH;             // 8,388,608 elements
  const size_t wsz   = (size_t)HH * HH;               // 1,048,576 elements
  unsigned short* qb  = u;
  unsigned short* kb  = u + plane;
  unsigned short* vb  = u + 2 * plane;
  unsigned short* yb  = u + 3 * plane;
  unsigned short* xb  = u + 4 * plane;
  unsigned short* wqb = u + 5 * plane;
  unsigned short* wkb = wqb + wsz;
  unsigned short* wvb = wkb + wsz;
  unsigned short* wob = wvb + wsz;

  cvt_bf16_kernel<<<dim3(1024), 256, 0, stream>>>(x,  xb,  (int)(plane / 2));
  cvt_bf16_kernel<<<dim3(256),  256, 0, stream>>>(Wq, wqb, (int)(wsz / 2));
  cvt_bf16_kernel<<<dim3(256),  256, 0, stream>>>(Wk, wkb, (int)(wsz / 2));
  cvt_bf16_kernel<<<dim3(256),  256, 0, stream>>>(Wv, wvb, (int)(wsz / 2));
  cvt_bf16_kernel<<<dim3(256),  256, 0, stream>>>(Wo, wob, (int)(wsz / 2));

  qkv_kernel<<<dim3(MTOK / 128, HH / 64, 3), 256, 0, stream>>>(
      xb, wqb, wkb, wvb, qb, kb, vb);
  scan_kernel<<<dim3(BB * NHH), 32, 0, stream>>>(
      qb, kb, vb, mask, theta, gate_w, gate_b, yb);
  out_ln_kernel<<<dim3(MTOK / 16), 256, 0, stream>>>(
      yb, wob, bo, x, gamma, beta, (float*)d_out);
}
